// Decoder_1408749273270
// MI455X (gfx1250) — matrix operand order repaired and checked by reading, not compiled, not run on hardware
//
#include <hip/hip_runtime.h>

// ---------------------------------------------------------------------------
// 4-layer LSTM, bsz=256, nhid=1024, nout=512, 128 steps.
// bf16 WMMA (v_wmma_f32_16x16x32_bf16) with fp32 accumulation, fused
// gates+cell per layer kernel, weights converted to bf16 once per call
// (59 MB -> L2 resident on MI455X's 192 MB L2), LDS-staged shared A operand,
// software-pipelined (register double-buffered) K loop, and 2 row-blocks per
// wave so each B (weight) fragment feeds 2 WMMAs (halves L2 weight traffic).
// ---------------------------------------------------------------------------

typedef __attribute__((ext_vector_type(16))) __bf16 v16bf;
typedef __attribute__((ext_vector_type(8)))  __bf16 v8bf;
typedef __attribute__((ext_vector_type(8)))  float  v8f;

#define NLAYERS 4
#define BSZ     256
#define NHID    1024
#define NOUT    512
#define STEPS   128
#define LDS_STRIDE (NHID + 8)   // +16B pad per row: b128 frag reads hit distinct banks

union BF16x16 { v16bf v; v8bf h[2]; };

__device__ __forceinline__ float sigf(float x) {
  return 1.0f / (1.0f + __expf(-x));
}

// A-matrix 16x32 bf16 fragment (wave32), contiguous k order (same as B):
//   lanes 0-15 : row M=lane,    K = k0 + {0..15}
//   lanes 16-31: row M=lane-16, K = k0 + {16..31}
__device__ __forceinline__ v16bf load_a_frag(const __bf16* A, int lane, int k0) {
  int row = lane & 15;
  int kb  = k0 + ((lane >> 4) << 4);
  const __bf16* p = A + row * LDS_STRIDE + kb;
  BF16x16 r;
  r.h[0] = *(const v8bf*)(p);
  r.h[1] = *(const v8bf*)(p + 8);
  return r.v;
}

// ---------------------------------------------------------------------------
// Pipelined GEMM pass over K=NHID: 4 gate tiles x 2 row-blocks per wave.
// B-matrix 32x16 bf16 fragment: column N = weight row (GEMM is h @ W^T, so a
// B column is a contiguous weight row along K).
//   lanes 0-15 : N = n0+lane,    K = k0..k0+15   (one 32B contiguous load)
//   lanes 16-31: N = n0+lane-16, K = k0+16..k0+31
// Next-iteration fragments are prefetched (wrap-around index, no guard
// branch) so s_wait only covers loads issued one full iteration earlier.
// Each B fragment is consumed by 2 WMMAs (row-blocks 0 and 1).
// ---------------------------------------------------------------------------
__device__ __forceinline__ void gemm_gates2(const __bf16* __restrict__ ldsA, // [32][LDS_STRIDE]
                                            const __bf16* __restrict__ Wm,
                                            int n0, int lane,
                                            v8f acc0[4], v8f acc1[4]) {
  const size_t gstride = (size_t)NHID * NHID;   // gate planes are NHID rows apart
  const __bf16* bp = Wm + (size_t)(n0 + (lane & 15)) * NHID + ((lane >> 4) << 4);
  const __bf16* ldsA1 = ldsA + 16 * LDS_STRIDE; // second row-block

  v16bf a0_c = load_a_frag(ldsA,  lane, 0);
  v16bf a1_c = load_a_frag(ldsA1, lane, 0);
  v16bf b_c[4];
  #pragma unroll
  for (int g = 0; g < 4; ++g) b_c[g] = *(const v16bf*)(bp + g * gstride);

  #pragma unroll 2
  for (int k0 = 0; k0 < NHID; k0 += 32) {
    const int kn = (k0 + 32) & (NHID - 1);   // wraps on last iter (value unused)
    v16bf a0_n = load_a_frag(ldsA,  lane, kn);
    v16bf a1_n = load_a_frag(ldsA1, lane, kn);
    v16bf b_n[4];
    #pragma unroll
    for (int g = 0; g < 4; ++g) b_n[g] = *(const v16bf*)(bp + g * gstride + kn);
    #pragma unroll
    for (int g = 0; g < 4; ++g)
      acc0[g] = __builtin_amdgcn_wmma_f32_16x16x32_bf16(
          false, a0_c, false, b_c[g], (short)0, acc0[g], false, false);
    #pragma unroll
    for (int g = 0; g < 4; ++g)
      acc1[g] = __builtin_amdgcn_wmma_f32_16x16x32_bf16(
          false, a1_c, false, b_c[g], (short)0, acc1[g], false, false);
    a0_c = a0_n;
    a1_c = a1_n;
    #pragma unroll
    for (int g = 0; g < 4; ++g) b_c[g] = b_n[g];
  }
}

// ---------------------------------------------------------------------------
// One LSTM layer for one step: gates = hprev @ U^T (+ xin @ W^T), then fused
// cell. grid = (8 row-pair-blocks, 8 col-groups), block = 256 (8 waves).
// Wave w owns hidden-col block n0 and computes the 4 gate tiles (columns
// n0 + g*NHID of the 4096-wide gate matrix) for 32 batch rows, then the cell.
// ---------------------------------------------------------------------------
__global__ __launch_bounds__(256)
void lstm_layer_kernel(const __bf16* __restrict__ hprev,  // [BSZ][NHID] this layer, prev step
                       const __bf16* __restrict__ xin,    // [BSZ][NHID] lower hy this step, or null
                       const __bf16* __restrict__ U,      // [4*NHID][NHID]
                       const __bf16* __restrict__ W,      // [4*NHID][NHID] or null
                       const float*  __restrict__ cprev,  // [BSZ][NHID] (in-place safe)
                       float*        __restrict__ cout,
                       __bf16*       __restrict__ hout)   // [BSZ][NHID]
{
  extern __shared__ char smem_raw[];
  __bf16* ldsH = (__bf16*)smem_raw;              // [32][LDS_STRIDE]
  __bf16* ldsX = ldsH + 32 * LDS_STRIDE;         // [32][LDS_STRIDE]

  const int m0   = blockIdx.x * 32;              // batch-row pair-block
  const int wave = threadIdx.x >> 5;
  const int lane = threadIdx.x & 31;
  const int n0   = (blockIdx.y * 8 + wave) * 16; // hidden-col block

  // Stage the shared A operand(s) into LDS (32 rows x 1024 K, b128 chunks).
  if (xin == nullptr) {
    for (int idx = threadIdx.x; idx < 32 * (NHID / 8); idx += 256) {
      int row = idx >> 7, cc = (idx & 127) << 3;
      *(v8bf*)(ldsH + row * LDS_STRIDE + cc) =
          *(const v8bf*)(hprev + (m0 + row) * NHID + cc);
    }
  } else {
    for (int idx = threadIdx.x; idx < 32 * (NHID / 8); idx += 256) {
      int row = idx >> 7, cc = (idx & 127) << 3;
      *(v8bf*)(ldsH + row * LDS_STRIDE + cc) =
          *(const v8bf*)(hprev + (m0 + row) * NHID + cc);
      *(v8bf*)(ldsX + row * LDS_STRIDE + cc) =
          *(const v8bf*)(xin + (m0 + row) * NHID + cc);
    }
  }

  // Preload the cell state early: its global latency hides under the GEMMs.
  // C/D 16x16 f32 layout: VGPR r -> row = rb*16 + r + ((lane>=16)?8:0),
  // col = n0 + (lane&15), for row-blocks rb = 0,1.
  const int col    = n0 + (lane & 15);
  const int rbase0 = m0 + ((lane >> 4) << 3);
  const int rbase1 = rbase0 + 16;
  float cp0[8], cp1[8];
  #pragma unroll
  for (int r = 0; r < 8; ++r) {
    cp0[r] = cprev[(rbase0 + r) * NHID + col];
    cp1[r] = cprev[(rbase1 + r) * NHID + col];
  }

  __syncthreads();

  v8f acc0[4] = {};   // gate order: i, f, c, o  (PyTorch fused layout)
  v8f acc1[4] = {};
  gemm_gates2(ldsH, U, n0, lane, acc0, acc1);
  if (W != nullptr) gemm_gates2(ldsX, W, n0, lane, acc0, acc1);

  // Fused LSTM cell on the accumulators (both row-blocks).
  #pragma unroll
  for (int r = 0; r < 8; ++r) {
    {
      int row  = rbase0 + r;
      float iv = sigf(acc0[0][r]);
      float fv = sigf(acc0[1][r]);
      float gv = tanhf(acc0[2][r]);
      float ov = sigf(acc0[3][r]);
      float cy = fv * cp0[r] + iv * gv;
      float hy = ov * tanhf(cy);
      cout[row * NHID + col] = cy;
      hout[row * NHID + col] = (__bf16)hy;
    }
    {
      int row  = rbase1 + r;
      float iv = sigf(acc1[0][r]);
      float fv = sigf(acc1[1][r]);
      float gv = tanhf(acc1[2][r]);
      float ov = sigf(acc1[3][r]);
      float cy = fv * cp1[r] + iv * gv;
      float hy = ov * tanhf(cy);
      cout[row * NHID + col] = cy;
      hout[row * NHID + col] = (__bf16)hy;
    }
  }
}

// ---------------------------------------------------------------------------
// Output projection: out = htop @ L^T  ([256,1024] x [1024,512] -> [256,512])
// grid = 64 blocks (16 row-blocks x 4 col-groups), 8 waves -> 512 tiles.
// Same pipelined K loop, single accumulator tile.
// ---------------------------------------------------------------------------
__global__ __launch_bounds__(256)
void proj_kernel(const __bf16* __restrict__ htop,  // [BSZ][NHID]
                 const __bf16* __restrict__ L,     // [NOUT][NHID]
                 float* __restrict__ out)          // [BSZ][NOUT]
{
  extern __shared__ char smem_raw[];
  __bf16* ldsH = (__bf16*)smem_raw;  // [16][LDS_STRIDE]

  const int rb   = blockIdx.x >> 2;
  const int cg   = blockIdx.x & 3;
  const int wave = threadIdx.x >> 5;
  const int lane = threadIdx.x & 31;
  const int m0   = rb * 16;
  const int n0   = (cg * 8 + wave) * 16;

  for (int idx = threadIdx.x; idx < 16 * (NHID / 8); idx += 256) {
    int row = idx >> 7, cc = (idx & 127) << 3;
    *(v8bf*)(ldsH + row * LDS_STRIDE + cc) =
        *(const v8bf*)(htop + (m0 + row) * NHID + cc);
  }
  __syncthreads();

  const __bf16* bp = L + (size_t)(n0 + (lane & 15)) * NHID + ((lane >> 4) << 4);

  v8f  acc = {};
  v16bf a_c = load_a_frag(ldsH, lane, 0);
  v16bf b_c = *(const v16bf*)(bp);
  #pragma unroll 2
  for (int k0 = 0; k0 < NHID; k0 += 32) {
    const int kn = (k0 + 32) & (NHID - 1);
    v16bf a_n = load_a_frag(ldsH, lane, kn);
    v16bf b_n = *(const v16bf*)(bp + kn);
    acc = __builtin_amdgcn_wmma_f32_16x16x32_bf16(
        false, a_c, false, b_c, (short)0, acc, false, false);
    a_c = a_n;
    b_c = b_n;
  }

  const int col   = n0 + (lane & 15);
  const int rbase = m0 + ((lane >> 4) << 3);
  #pragma unroll
  for (int r = 0; r < 8; ++r)
    out[(rbase + r) * NOUT + col] = acc[r];
}

// ---------------------------------------------------------------------------
// fp32 -> bf16 (RNE) conversion, grid-stride.
// ---------------------------------------------------------------------------
__global__ void f32_to_bf16_kernel(const float* __restrict__ src,
                                   __bf16* __restrict__ dst, int n) {
  int i      = blockIdx.x * blockDim.x + threadIdx.x;
  int stride = gridDim.x * blockDim.x;
  for (; i < n; i += stride) dst[i] = (__bf16)src[i];
}

// ---------------------------------------------------------------------------
extern "C" void kernel_launch(void* const* d_in, const int* in_sizes, int n_in,
                              void* d_out, int out_size, void* d_ws, size_t ws_size,
                              hipStream_t stream) {
  const float* h0  = (const float*)d_in[0];  // [4][256][1024]
  const float* c0  = (const float*)d_in[1];  // [4][256][1024]
  const float* U_w = (const float*)d_in[2];  // [4][4096][1024]
  const float* W_w = (const float*)d_in[3];  // [3][4096][1024]
  const float* L_w = (const float*)d_in[4];  // [512][1024]

  // Carve workspace (everything re-initialized every call; no cross-call state).
  char* ws = (char*)d_ws;
  size_t off = 0;
  auto carve = [&](size_t bytes) -> void* {
    void* p = ws + off;
    off = (off + bytes + 255) & ~(size_t)255;
    return p;
  };
  const size_t US = (size_t)4 * NHID * NHID;       // per-layer U slice (elems)
  const size_t HS = (size_t)BSZ * NHID;            // per-layer h/c slice (elems)

  __bf16* Ub = (__bf16*)carve((size_t)NLAYERS * US * 2);
  __bf16* Wb = (__bf16*)carve((size_t)(NLAYERS - 1) * US * 2);
  __bf16* Lb = (__bf16*)carve((size_t)NOUT * NHID * 2);
  __bf16* hA = (__bf16*)carve((size_t)NLAYERS * HS * 2);
  __bf16* hB = (__bf16*)carve((size_t)NLAYERS * HS * 2);
  float*  cS = (float*)carve((size_t)NLAYERS * HS * 4);

  auto cvt = [&](const float* s, __bf16* d, int n) {
    f32_to_bf16_kernel<<<dim3(1024), dim3(256), 0, stream>>>(s, d, n);
  };
  cvt(U_w, Ub, (int)(NLAYERS * US));
  cvt(W_w, Wb, (int)((NLAYERS - 1) * US));
  cvt(L_w, Lb, NOUT * NHID);
  cvt(h0,  hA, (int)(NLAYERS * HS));
  hipMemcpyAsync(cS, c0, (size_t)NLAYERS * HS * sizeof(float),
                 hipMemcpyDeviceToDevice, stream);

  float* out = (float*)d_out;
  const size_t layerLds = (size_t)2 * 32 * LDS_STRIDE * sizeof(__bf16);  // ~132 KB
  const size_t projLds  = (size_t)16 * LDS_STRIDE * sizeof(__bf16);

  // out[0]: projection of the initial top-layer hidden state.
  proj_kernel<<<dim3(64), dim3(256), projLds, stream>>>(hA + 3 * HS, Lb, out);

  __bf16* hp = hA;  // previous-step h (read)
  __bf16* hc = hB;  // current-step h (written; also xin chain within the step)
  for (int t = 1; t < STEPS; ++t) {
    for (int l = 0; l < NLAYERS; ++l) {
      lstm_layer_kernel<<<dim3(8, 8), dim3(256), layerLds, stream>>>(
          hp + l * HS,
          l ? hc + (l - 1) * HS : nullptr,
          Ub + l * US,
          l ? Wb + (l - 1) * US : nullptr,
          cS + l * HS, cS + l * HS,   // c updates in-place (tile-local)
          hc + l * HS);
    }
    proj_kernel<<<dim3(64), dim3(256), projLds, stream>>>(
        hc + 3 * HS, Lb, out + (size_t)t * BSZ * NOUT);
    __bf16* tmp = hp; hp = hc; hc = tmp;
  }
}
